// FouriDown_19670950216197
// MI455X (gfx1250) — matrix operand order repaired and checked by reading, not compiled, not run on hardware
//
#include <hip/hip_runtime.h>
#include <hip/hip_bf16.h>
#include <math.h>

typedef __attribute__((ext_vector_type(16))) _Float16 v16h;
typedef __attribute__((ext_vector_type(8)))  float    v8f;

#define TWO_PI 6.2831853071795864769f

// ---------------------------------------------------------------------------
// Generic in-LDS radix-2 Stockham FFT (autosorting, no bit reversal).
// sign = -1 forward, +1 inverse (unnormalized). Result pointer returned.
// All threads of the block must call with the same N.
// ---------------------------------------------------------------------------
template<int N>
__device__ inline float2* fft_stockham(float2* buf0, float2* buf1,
                                       int tid, int nthreads, float sign)
{
  float2* src = buf0;
  float2* dst = buf1;
  #pragma unroll 1
  for (int m = 1; m < N; m <<= 1) {
    __syncthreads();
    for (int i = tid; i < N/2; i += nthreads) {
      const int k = i & (m - 1);
      const float2 c0 = src[i];
      const float2 c1 = src[i + N/2];
      const float ang = sign * (TWO_PI / (float)N) * (float)(i - k);
      float sw, cw;
      __sincosf(ang, &sw, &cw);
      const float sx = c0.x - c1.x;
      const float sy = c0.y - c1.y;
      const int j = 2*i - k;
      dst[j]     = make_float2(c0.x + c1.x, c0.y + c1.y);
      dst[j + m] = make_float2(cw*sx - sw*sy, cw*sy + sw*sx);
    }
    float2* t = src; src = dst; dst = t;
  }
  __syncthreads();
  return src;
}

// ---------------------------------------------------------------------------
// Kernel 1: the only part of the forward 2D FFT the reference consumes:
//   sc[b*8 + w]     = Re F(b, c=0, h=0, w)   (w = 0..3)
//   sc[b*8 + 4 + w] = Im F(b, c=0, h=0, w)
// F(b,0,0,w) = sum_{w'} colsum[w'] * exp(-2*pi*i*w*w'/512)
// ---------------------------------------------------------------------------
__global__ __launch_bounds__(256) void k_scalars(const float* __restrict__ x,
                                                 float* __restrict__ sc)
{
  __shared__ float colsum[512];
  __shared__ float red[256];
  const int b   = blockIdx.x;
  const int tid = threadIdx.x;
  const float* xp = x + (size_t)b * 32u * 512u * 512u;   // channel 0 plane
  for (int w = tid; w < 512; w += 256) {
    float s = 0.0f;
    for (int h = 0; h < 512; ++h) {
      __builtin_prefetch(xp + (size_t)(h + 8)*512u + (unsigned)w, 0, 3);
      s += xp[(size_t)h*512u + (unsigned)w];
    }
    colsum[w] = s;
  }
  __syncthreads();

  float accR[4] = {0,0,0,0}, accI[4] = {0,0,0,0};
  for (int wp = tid; wp < 512; wp += 256) {
    const float v = colsum[wp];
    #pragma unroll
    for (int w = 0; w < 4; ++w) {
      const float ang = -(TWO_PI/512.0f) * (float)(w * wp);
      float s, c;
      __sincosf(ang, &s, &c);
      accR[w] += v * c;
      accI[w] += v * s;
    }
  }
  #pragma unroll 1
  for (int w = 0; w < 8; ++w) {
    const float v = (w < 4) ? accR[w & 3] : accI[w & 3];
    __syncthreads();
    red[tid] = v;
    __syncthreads();
    for (int off = 128; off > 0; off >>= 1) {
      if (tid < off) red[tid] += red[tid + off];
      __syncthreads();
    }
    if (tid == 0) sc[b*8 + w] = red[0];
  }
}

// ---------------------------------------------------------------------------
// A-fragment builder for V_WMMA_F32_16X16X32_F16 (16-bit A is 16x32, K padded
// to 32 with zeros). Layout used here: lane l (hi=l>>4, M=l&15); vector
// element t maps to K = t + 16*hi. The 16x16 weight tile is
// block-diagonal with four 4x4 group blocks: rows=out channels, cols=in.
// ---------------------------------------------------------------------------
__device__ inline v16h make_a_frag(const float* __restrict__ w, int g0, int lane)
{
  const int hi = lane >> 4;
  const int M  = lane & 15;
  v16h a;
  #pragma unroll
  for (int t = 0; t < 16; ++t) {
    const int K = t + 16*hi;
    float v = 0.0f;
    if (K < 16 && ((M >> 2) == (K >> 2)))
      v = w[(g0 + (M >> 2))*16 + (M & 3)*4 + (K & 3)];
    a[t] = (_Float16)v;
  }
  return a;
}

// ---------------------------------------------------------------------------
// Kernel 2: fuse (gather + grouped conv x2 + leaky + softmax + complex mix).
// 256 threads = 8 waves; wave r owns channel tile [16r,16r+16) = 4 conv groups.
// Conv: D = Wtile(16x16, block-diag) * X(16x16 pixels), via one f16 WMMA per
// layer (K zero-padded 16->32), f32 accumulate seeded with bias.
// B layout used here (16-bit, 32x16): lane n=l&15, element t -> K = t + 16*(l>>4); our
// K<16 data lives entirely in lanes 0..15, upper half stays zero.
// C/D layout (f32): lane l holds column n=l&15, rows q + 8*(l>>4).
// ---------------------------------------------------------------------------
#define NITER 16

__global__ __launch_bounds__(256) void k_fuse(
    const float* __restrict__ sc,
    const int*   __restrict__ eidx,
    const float* __restrict__ w1r, const float* __restrict__ b1r,
    const float* __restrict__ w2r, const float* __restrict__ b2r,
    const float* __restrict__ w1i, const float* __restrict__ b1i,
    const float* __restrict__ w2i, const float* __restrict__ b2i,
    float* __restrict__ Gr, float* __restrict__ Gi)
{
  __shared__ float sWr[128*16];
  __shared__ float sWi[128*16];
  __shared__ float sFr[128*16];
  __shared__ float sFi[128*16];
  __shared__ _Float16 sY[8][16*16];

  const int lane = threadIdx.x & 31;
  const int wv   = threadIdx.x >> 5;   // tile row r
  const int hi   = lane >> 4;
  const int n    = lane & 15;
  const int g0   = wv * 4;             // first conv group of this tile

  // Weight fragments, loaded once per block (chunk-invariant).
  const v16h a1r = make_a_frag(w1r, g0, lane);
  const v16h a2r = make_a_frag(w2r, g0, lane);
  const v16h a1i = make_a_frag(w1i, g0, lane);
  const v16h a2i = make_a_frag(w2i, g0, lane);

  float bias1r[8], bias2r[8], bias1i[8], bias2i[8];
  #pragma unroll
  for (int q = 0; q < 8; ++q) {
    const int row = q + 8*hi;
    const int gg = g0 + (row >> 2), oo = row & 3;
    bias1r[q] = b1r[gg*4 + oo];
    bias2r[q] = b2r[gg*4 + oo];
    bias1i[q] = b1i[gg*4 + oo];
    bias2i[q] = b2i[gg*4 + oo];
  }

  #pragma unroll 1
  for (int it = 0; it < NITER; ++it) {
    const int chunk = blockIdx.x * NITER + it;
    const int b     = chunk >> 12;            // 4096 16-pixel chunks / batch
    const int pbase = (chunk & 4095) * 16;

    float scR[4], scI[4];
    #pragma unroll
    for (int q = 0; q < 4; ++q) {
      scR[q] = sc[b*8 + q];
      scI[q] = sc[b*8 + 4 + q];
    }

    // Gather fuse values (each is one of 4 scalars) -> B fragments + f32 LDS
    v16h br = {}, bimag = {};
    if (hi == 0) {
      #pragma unroll
      for (int t = 0; t < 16; ++t) {
        const int v  = wv*16 + t;                         // channel u = p*32+c
        const int id = eidx[(size_t)v*65536u + (unsigned)(pbase + n)] & 3;
        const float fr = scR[id];
        const float fi = scI[id];
        sFr[v*16 + n] = fr;
        sFi[v*16 + n] = fi;
        br[t]    = (_Float16)fr;
        bimag[t] = (_Float16)fi;
      }
    }

    // ---------------- real branch ----------------
    {
      v8f acc;
      #pragma unroll
      for (int q = 0; q < 8; ++q) acc[q] = bias1r[q];
      acc = __builtin_amdgcn_wmma_f32_16x16x32_f16(false, a1r, false, br,
                                                   (short)0, acc, false, false);
      #pragma unroll
      for (int q = 0; q < 8; ++q) { const float v = acc[q]; acc[q] = (v > 0.0f) ? v : 0.1f*v; }
      #pragma unroll
      for (int q = 0; q < 8; ++q) sY[wv][(q + 8*hi)*16 + n] = (_Float16)acc[q];
      v16h y = {};
      if (hi == 0) {
        #pragma unroll
        for (int t = 0; t < 16; ++t) y[t] = sY[wv][t*16 + n];
      }
      v8f acc2;
      #pragma unroll
      for (int q = 0; q < 8; ++q) acc2[q] = bias2r[q];
      acc2 = __builtin_amdgcn_wmma_f32_16x16x32_f16(false, a2r, false, y,
                                                    (short)0, acc2, false, false);
      #pragma unroll
      for (int q = 0; q < 8; ++q) sWr[(wv*16 + q + 8*hi)*16 + n] = acc2[q];
    }

    // ---------------- imag branch ----------------
    {
      v8f acc;
      #pragma unroll
      for (int q = 0; q < 8; ++q) acc[q] = bias1i[q];
      acc = __builtin_amdgcn_wmma_f32_16x16x32_f16(false, a1i, false, bimag,
                                                   (short)0, acc, false, false);
      #pragma unroll
      for (int q = 0; q < 8; ++q) { const float v = acc[q]; acc[q] = (v > 0.0f) ? v : 0.1f*v; }
      #pragma unroll
      for (int q = 0; q < 8; ++q) sY[wv][(q + 8*hi)*16 + n] = (_Float16)acc[q];
      v16h y = {};
      if (hi == 0) {
        #pragma unroll
        for (int t = 0; t < 16; ++t) y[t] = sY[wv][t*16 + n];
      }
      v8f acc2;
      #pragma unroll
      for (int q = 0; q < 8; ++q) acc2[q] = bias2i[q];
      acc2 = __builtin_amdgcn_wmma_f32_16x16x32_f16(false, a2i, false, y,
                                                    (short)0, acc2, false, false);
      #pragma unroll
      for (int q = 0; q < 8; ++q) sWi[(wv*16 + q + 8*hi)*16 + n] = acc2[q];
    }

    __syncthreads();

    // Softmax over the 4 patch slots (couples tile rows) + complex mixing.
    #pragma unroll
    for (int half = 0; half < 2; ++half) {
      const int item = (int)threadIdx.x + half*256;  // 512 = 32 c * 16 pixels
      const int c  = item >> 4;
      const int nn = item & 15;
      float wr4[4], wi4[4];
      #pragma unroll
      for (int p = 0; p < 4; ++p) {
        const int v = p*32 + c;
        wr4[p] = sWr[v*16 + nn] + 0.25f;
        wi4[p] = sWi[v*16 + nn] + 0.25f;
      }
      const float mr = fmaxf(fmaxf(wr4[0], wr4[1]), fmaxf(wr4[2], wr4[3]));
      const float mi = fmaxf(fmaxf(wi4[0], wi4[1]), fmaxf(wi4[2], wi4[3]));
      float er[4], ei[4], sumr = 0.0f, sumi = 0.0f;
      #pragma unroll
      for (int p = 0; p < 4; ++p) {
        er[p] = __expf(wr4[p] - mr); sumr += er[p];
        ei[p] = __expf(wi4[p] - mi); sumi += ei[p];
      }
      const float invr = 1.0f / sumr, invi = 1.0f / sumi;
      float outR = 0.0f, outI = 0.0f;
      #pragma unroll
      for (int p = 0; p < 4; ++p) {
        const int v = p*32 + c;
        const float fr = sFr[v*16 + nn];
        const float fi = sFi[v*16 + nn];
        const float srw = er[p] * invr;
        const float siw = ei[p] * invi;
        outR += fr*srw - fi*siw;
        outI += fr*siw + fi*srw;
      }
      const size_t go = ((size_t)(b*32 + c))*65536u + (unsigned)(pbase + nn);
      Gr[go] = outR;
      Gi[go] = outI;
    }
    __syncthreads();
  }
}

// ---------------------------------------------------------------------------
// Kernel 3: inverse FFT along width (256-point) for every (b,c,hs) row.
// ---------------------------------------------------------------------------
__global__ __launch_bounds__(256) void k_ifft_rows(const float* __restrict__ Gr,
                                                   const float* __restrict__ Gi,
                                                   float* __restrict__ Tr,
                                                   float* __restrict__ Ti)
{
  __shared__ float2 bufA[4][256];
  __shared__ float2 bufB[4][256];
  const int sub = threadIdx.x >> 6;
  const int tid = threadIdx.x & 63;
  const int line = blockIdx.x*4 + sub;        // (b*32 + c)*256 + hs
  const size_t base = (size_t)line * 256u;
  for (int i = tid; i < 256; i += 64)
    bufA[sub][i] = make_float2(Gr[base + i], Gi[base + i]);
  float2* res = fft_stockham<256>(bufA[sub], bufB[sub], tid, 64, +1.0f);
  for (int i = tid; i < 256; i += 64) {
    Tr[base + i] = res[i].x;
    Ti[base + i] = res[i].y;
  }
}

// ---------------------------------------------------------------------------
// Kernel 4: inverse FFT along height + |.|/65536 + fused 2x2 avg-pool of x.
// ---------------------------------------------------------------------------
__global__ __launch_bounds__(256) void k_ifft_cols_finish(const float* __restrict__ Tr,
                                                          const float* __restrict__ Ti,
                                                          const float* __restrict__ x,
                                                          float* __restrict__ out)
{
  __shared__ float2 bufA[4][256];
  __shared__ float2 bufB[4][256];
  const int sub = threadIdx.x >> 6;
  const int tid = threadIdx.x & 63;
  const int colg  = blockIdx.x*4 + sub;
  const int plane = colg >> 8;                // b*32 + c
  const int w     = colg & 255;
  const size_t base = (size_t)plane * 65536u + (unsigned)w;
  for (int i = tid; i < 256; i += 64)
    bufA[sub][i] = make_float2(Tr[base + (size_t)i*256u], Ti[base + (size_t)i*256u]);
  float2* res = fft_stockham<256>(bufA[sub], bufB[sub], tid, 64, +1.0f);
  const size_t xplane = (size_t)plane * 512u * 512u;
  for (int i = tid; i < 256; i += 64) {
    const float re = res[i].x, im = res[i].y;
    const float mag = sqrtf(re*re + im*im) * (1.0f/65536.0f);
    const size_t xb = xplane + (size_t)(2*i)*512u + (unsigned)(2*w);
    const float pooled = 0.25f*(x[xb] + x[xb + 1] + x[xb + 512] + x[xb + 513]);
    out[base + (size_t)i*256u] = mag + pooled;
  }
}

// ---------------------------------------------------------------------------
// Host launch. Workspace layout (floats):
//   [0..16)            : forward-FFT scalars (8 per batch)
//   [64 .. 64+NG)      : Gr
//   [.. +NG)           : Gi
//   [.. +NG)           : Tr
//   [.. +NG)           : Ti          (NG = B*C*Hs*Ws = 4,194,304 -> ~64 MB total)
// ---------------------------------------------------------------------------
extern "C" void kernel_launch(void* const* d_in, const int* in_sizes, int n_in,
                              void* d_out, int out_size, void* d_ws, size_t ws_size,
                              hipStream_t stream)
{
  const float* x    = (const float*)d_in[0];
  const int*   eidx = (const int*)  d_in[1];
  const float* w1r  = (const float*)d_in[2];
  const float* b1r  = (const float*)d_in[3];
  const float* w2r  = (const float*)d_in[4];
  const float* b2r  = (const float*)d_in[5];
  const float* w1i  = (const float*)d_in[6];
  const float* b1i  = (const float*)d_in[7];
  const float* w2i  = (const float*)d_in[8];
  const float* b2i  = (const float*)d_in[9];
  float* out = (float*)d_out;
  float* ws  = (float*)d_ws;

  const size_t NG = (size_t)2 * 32 * 256 * 256;  // 4,194,304
  float* sc = ws;
  float* Gr = ws + 64;
  float* Gi = Gr + NG;
  float* Tr = Gi + NG;
  float* Ti = Tr + NG;

  k_scalars<<<2, 256, 0, stream>>>(x, sc);
  k_fuse<<<512, 256, 0, stream>>>(sc, eidx, w1r, b1r, w2r, b2r,
                                  w1i, b1i, w2i, b2i, Gr, Gi);
  k_ifft_rows<<<4096, 256, 0, stream>>>(Gr, Gi, Tr, Ti);
  k_ifft_cols_finish<<<4096, 256, 0, stream>>>(Tr, Ti, x, out);

  (void)in_sizes; (void)n_in; (void)out_size; (void)ws_size;
}
